// SelfAttention_47244640256234
// MI455X (gfx1250) — matrix operand order repaired and checked by reading, not compiled, not run on hardware
//
#include <hip/hip_runtime.h>
#include <math.h>
#include <stdint.h>

// Problem constants (from reference): B=32, S=1024, D=768, all fp32 I/O.
#define BB 32
#define SS 1024
#define DD 768

typedef __bf16 bf16_t;
typedef __attribute__((ext_vector_type(8)))  __bf16 v8bf;
typedef __attribute__((ext_vector_type(16))) __bf16 v16bf;
typedef __attribute__((ext_vector_type(8)))  float  v8f;

union V16U { v16bf v; v8bf h[2]; };

// ---------------------------------------------------------------------------
// Kernel 1: H = bf16(x + pos)   (grid: [S*D/256, B])
// ---------------------------------------------------------------------------
__global__ __launch_bounds__(256) void addpos_cvt_kernel(
    const float* __restrict__ x, const float* __restrict__ pos,
    bf16_t* __restrict__ H) {
  const long r = (long)blockIdx.x * 256 + threadIdx.x;   // 0 .. S*D-1
  const long b = blockIdx.y;
  H[b * (long)(SS * DD) + r] = (bf16_t)(x[b * (long)(SS * DD) + r] + pos[r]);
}

// ---------------------------------------------------------------------------
// Kernel 2: Wb = bf16({Wq,Wk,Wv})   (grid: [D*D/256, 3])
// ---------------------------------------------------------------------------
__global__ __launch_bounds__(256) void cvtw_kernel(
    const float* __restrict__ Wq, const float* __restrict__ Wk,
    const float* __restrict__ Wv, bf16_t* __restrict__ Wb) {
  const int w = blockIdx.y;
  const float* src = (w == 0) ? Wq : ((w == 1) ? Wk : Wv);
  const long i = (long)blockIdx.x * 256 + threadIdx.x;
  Wb[(long)w * (DD * DD) + i] = (bf16_t)src[i];
}

// ---------------------------------------------------------------------------
// Tiled bf16 WMMA GEMM, 256 threads (8 waves), C-tile 128x128, Kstep 32.
// Wave w owns rows (w&3)*32..+31, cols (w>>2)*64..+63  (8 WMMAs / K-step).
// Global->LDS staging uses GLOBAL_LOAD_ASYNC_TO_LDS_B128 (ASYNCcnt) with
// double-buffered LDS tiles: tile i+1 is issued before computing tile i.
//   C = A[M x Kdim] (row-major) * Bt[N x Kdim]^T (Bt rows = B columns)
// MODE 0: bf16 out, row-major [.. x DD], + bias        (Q, K projections)
// MODE 1: bf16 out, TRANSPOSED [DD x SS], + bias       (V -> V^T)
// MODE 2: f32 out, row-major [.. x SS], * scale        (scores)
// MODE 3: f32 out, row-major [.. x DD]                 (final output)
// ---------------------------------------------------------------------------
template <int MODE>
__global__ __launch_bounds__(256) void gemm128_kernel(
    const bf16_t* __restrict__ A, long aBatch,
    const bf16_t* __restrict__ Bt, long bBatch,
    const float* __restrict__ bias, float scale,
    void* __restrict__ OutP, long oBatch, int Kdim) {
  constexpr int LDT = 40;           // 32 + 8 pad bf16 (80B rows, 16B aligned)
  constexpr int TILE = 128 * LDT;   // one 128x32 tile (padded)
  __shared__ bf16_t As[2 * TILE];
  __shared__ bf16_t Bs[2 * TILE];

  const int b  = blockIdx.z;
  const int m0 = blockIdx.x * 128;
  const int n0 = blockIdx.y * 128;
  const bf16_t* Ab = A  + (long)b * aBatch;
  const bf16_t* Bb = Bt + (long)b * bBatch;

  const int t = threadIdx.x;
  const int lane = t & 31;
  const int wave = t >> 5;
  const int khalf = lane >> 4;          // 0: lanes 0-15, 1: lanes 16-31
  const int wm = (wave & 3) * 32;       // wave row base within C tile
  const int wn = (wave >> 2) * 64;      // wave col base within C tile

  // Per-thread staging chunks (16B each): 128 rows x 4 chunks, 2 per matrix.
  const int row0 = t >> 2;              // 0..63
  const int off0 = (t & 3) * 8;         // bf16 elements
  // second chunk is row0+64, same off0

  v8f acc[2][4] = {};

  // Issue async loads of the 128x32 A and Bt tiles for K-offset kt into buf.
  auto issue = [&](int kt, int buf) {
    bf16_t* dA = As + buf * TILE;
    bf16_t* dB = Bs + buf * TILE;
    // Flat LDS pointer low 32 bits == LDS byte address (aperture rule 10.2).
    const unsigned la0 = (unsigned)(uintptr_t)(dA + row0 * LDT + off0);
    const unsigned la1 = (unsigned)(uintptr_t)(dA + (row0 + 64) * LDT + off0);
    const unsigned lb0 = (unsigned)(uintptr_t)(dB + row0 * LDT + off0);
    const unsigned lb1 = (unsigned)(uintptr_t)(dB + (row0 + 64) * LDT + off0);
    const bf16_t* ga0 = Ab + (long)(m0 + row0) * Kdim + kt + off0;
    const bf16_t* ga1 = Ab + (long)(m0 + row0 + 64) * Kdim + kt + off0;
    const bf16_t* gb0 = Bb + (long)(n0 + row0) * Kdim + kt + off0;
    const bf16_t* gb1 = Bb + (long)(n0 + row0 + 64) * Kdim + kt + off0;
    asm volatile("global_load_async_to_lds_b128 %0, %1, off"
                 :: "v"(la0), "v"(ga0) : "memory");
    asm volatile("global_load_async_to_lds_b128 %0, %1, off"
                 :: "v"(la1), "v"(ga1) : "memory");
    asm volatile("global_load_async_to_lds_b128 %0, %1, off"
                 :: "v"(lb0), "v"(gb0) : "memory");
    asm volatile("global_load_async_to_lds_b128 %0, %1, off"
                 :: "v"(lb1), "v"(gb1) : "memory");
  };

  issue(0, 0);
  const int nsteps = Kdim >> 5;
  for (int i = 0; i < nsteps; ++i) {
    const bf16_t* Asb = As + (i & 1) * TILE;
    const bf16_t* Bsb = Bs + (i & 1) * TILE;
    const bool more = (i + 1) < nsteps;
    if (more) {
      // WAR on buffer (i+1)&1 is safe: last readers finished at the trailing
      // barrier of iteration i-1.
      issue((i + 1) * 32, (i + 1) & 1);
      asm volatile("s_wait_asynccnt 0x4" ::: "memory");  // tile i complete
    } else {
      asm volatile("s_wait_asynccnt 0x0" ::: "memory");
    }
    __syncthreads();

    // A fragments (16x32 bf16): lane holds row M; lanes 0-15: K 0-7 / 16-23,
    // lanes 16-31: K 8-15 / 24-31.
    V16U af[2];
#pragma unroll
    for (int a = 0; a < 2; ++a) {
      const int r = wm + a * 16 + (lane & 15);
      af[a].h[0] = *reinterpret_cast<const v8bf*>(Asb + r * LDT + khalf * 8);
      af[a].h[1] = *reinterpret_cast<const v8bf*>(Asb + r * LDT + 16 + khalf * 8);
    }
#pragma unroll
    for (int nt = 0; nt < 4; ++nt) {
      // B fragment (32x16): lane n holds column N=n; lanes 0-15 K=0-7 / 16-23,
      // lanes 16-31 K=8-15 / 24-31; Bs holds B^T rows.
      const int brow = wn + nt * 16 + (lane & 15);
      V16U bfr;
      bfr.h[0] = *reinterpret_cast<const v8bf*>(Bsb + brow * LDT + khalf * 8);
      bfr.h[1] = *reinterpret_cast<const v8bf*>(Bsb + brow * LDT + 16 + khalf * 8);
#pragma unroll
      for (int a = 0; a < 2; ++a)
        acc[a][nt] = __builtin_amdgcn_wmma_f32_16x16x32_bf16(
            false, af[a].v, false, bfr.v, (short)0, acc[a][nt], false, false);
    }
    __syncthreads();  // all waves done reading buf (i&1) before it is refilled
  }

  // C/D layout: VGPR j holds M = j (lanes 0-15) or j+8 (lanes 16-31), N = lane&15.
  const int ncol = lane & 15;
#pragma unroll
  for (int a = 0; a < 2; ++a) {
    const int mbase = m0 + wm + a * 16 + khalf * 8;
    if (MODE == 0) {  // bf16 row-major + bias
      bf16_t* O = (bf16_t*)OutP + (long)b * oBatch;
#pragma unroll
      for (int nt = 0; nt < 4; ++nt) {
        const int n = n0 + wn + nt * 16 + ncol;
        const float bv = bias[n];
#pragma unroll
        for (int j = 0; j < 8; ++j)
          O[(long)(mbase + j) * DD + n] = (bf16_t)(acc[a][nt][j] + bv);
      }
    } else if (MODE == 1) {  // bf16 transposed (V^T[D][S]) + bias, b128 stores
      bf16_t* O = (bf16_t*)OutP + (long)b * oBatch;
#pragma unroll
      for (int nt = 0; nt < 4; ++nt) {
        const int n = n0 + wn + nt * 16 + ncol;
        const float bv = bias[n];
        v8bf pk;
#pragma unroll
        for (int j = 0; j < 8; ++j) pk[j] = (bf16_t)(acc[a][nt][j] + bv);
        *reinterpret_cast<v8bf*>(O + (long)n * SS + mbase) = pk;  // 16B aligned
      }
    } else if (MODE == 2) {  // f32 scores * scale, row stride SS
      float* O = (float*)OutP + (long)b * oBatch;
#pragma unroll
      for (int nt = 0; nt < 4; ++nt) {
        const int n = n0 + wn + nt * 16 + ncol;
#pragma unroll
        for (int j = 0; j < 8; ++j)
          O[(long)(mbase + j) * SS + n] = acc[a][nt][j] * scale;
      }
    } else {  // MODE 3: f32 output, row stride DD
      float* O = (float*)OutP + (long)b * oBatch;
#pragma unroll
      for (int nt = 0; nt < 4; ++nt) {
        const int n = n0 + wn + nt * 16 + ncol;
#pragma unroll
        for (int j = 0; j < 8; ++j)
          O[(long)(mbase + j) * DD + n] = acc[a][nt][j];
      }
    }
  }
}

// ---------------------------------------------------------------------------
// Row softmax, wave32 per row, 1024 f32 in registers (32/lane).
// Writes bf16 P IN PLACE over the f32 scores, P row r on half of score row r/2 (reads complete before writes
// within the lockstep wave; for r > 0 that score row is read by another wave).
// ---------------------------------------------------------------------------
__global__ __launch_bounds__(128) void softmax_kernel(
    float* __restrict__ Sc, bf16_t* __restrict__ P) {
  const int lane = threadIdx.x & 31;
  const long row = (long)blockIdx.x * 4 + (threadIdx.x >> 5);
  const float* src = Sc + row * SS;
  float v[32];
  float mx = -3.402823466e38f;
#pragma unroll
  for (int i = 0; i < 32; ++i) {
    v[i] = src[lane + 32 * i];
    mx = fmaxf(mx, v[i]);
  }
#pragma unroll
  for (int o = 16; o > 0; o >>= 1) mx = fmaxf(mx, __shfl_xor(mx, o, 32));
  float sum = 0.f;
#pragma unroll
  for (int i = 0; i < 32; ++i) {
    v[i] = __expf(v[i] - mx);
    sum += v[i];
  }
#pragma unroll
  for (int o = 16; o > 0; o >>= 1) sum += __shfl_xor(sum, o, 32);
  const float inv = 1.0f / sum;
  bf16_t* dst = P + row * SS;
#pragma unroll
  for (int i = 0; i < 32; ++i) dst[lane + 32 * i] = (bf16_t)(v[i] * inv);
}

// ---------------------------------------------------------------------------
// Workspace layout (bytes), ~272 MB total:
//   [0,           50331648)  Q   bf16 [B][S][D]
//   [50331648,   100663296)  K   bf16 [B][S][D]
//   [100663296,  150994944)  V^T bf16 [B][D][S]
//   [150994944,  201326592)  H   bf16 [B][S][D]   (dead after projections)
//   [201326592,  204865536)  Wb  bf16 3x[D][D]    (dead after projections)
//   [150994944,  285212672)  Sc  f32  [B][S][S]   (reuses H/Wb region;
//                                                  softmax rewrites it as bf16 P)
// ---------------------------------------------------------------------------
extern "C" void kernel_launch(void* const* d_in, const int* in_sizes, int n_in,
                              void* d_out, int out_size, void* d_ws,
                              size_t ws_size, hipStream_t stream) {
  const float* x   = (const float*)d_in[0];
  const float* Wq  = (const float*)d_in[1];
  const float* bq  = (const float*)d_in[2];
  const float* Wk  = (const float*)d_in[3];
  const float* bk  = (const float*)d_in[4];
  const float* Wv  = (const float*)d_in[5];
  const float* bv  = (const float*)d_in[6];
  const float* pos = (const float*)d_in[7];

  char* ws = (char*)d_ws;
  bf16_t* Qb  = (bf16_t*)(ws + 0L);
  bf16_t* Kb  = (bf16_t*)(ws + 50331648L);
  bf16_t* VTb = (bf16_t*)(ws + 100663296L);
  bf16_t* Hb  = (bf16_t*)(ws + 150994944L);
  bf16_t* Wb  = (bf16_t*)(ws + 201326592L);
  float*  Sc  = (float*)(ws + 150994944L);  // aliases Hb/Wb (stream-ordered)
  bf16_t* Pb  = (bf16_t*)Sc;                // in-place after softmax

  const float scale = 1.0f / sqrtf((float)DD);
  const long sdBatch = (long)SS * DD;   // 786432
  const long ssBatch = (long)SS * SS;   // 1048576

  addpos_cvt_kernel<<<dim3((SS * DD) / 256, BB), 256, 0, stream>>>(x, pos, Hb);
  cvtw_kernel<<<dim3((DD * DD) / 256, 3), 256, 0, stream>>>(Wq, Wk, Wv, Wb);

  // Q = H*Wq^T + bq ; K = H*Wk^T + bk   (W stored [out,in] => Bt rows = W rows)
  gemm128_kernel<0><<<dim3(SS / 128, DD / 128, BB), 256, 0, stream>>>(
      Hb, sdBatch, Wb, 0L, bq, 1.f, Qb, sdBatch, DD);
  gemm128_kernel<0><<<dim3(SS / 128, DD / 128, BB), 256, 0, stream>>>(
      Hb, sdBatch, Wb + (long)DD * DD, 0L, bk, 1.f, Kb, sdBatch, DD);
  // V^T = (H*Wv^T + bv)^T
  gemm128_kernel<1><<<dim3(SS / 128, DD / 128, BB), 256, 0, stream>>>(
      Hb, sdBatch, Wb + 2L * DD * DD, 0L, bv, 1.f, VTb, sdBatch, DD);
  // Sc = scale * Q*K^T
  gemm128_kernel<2><<<dim3(SS / 128, SS / 128, BB), 256, 0, stream>>>(
      Qb, sdBatch, Kb, sdBatch, nullptr, scale, Sc, ssBatch, DD);
  // P = softmax(Sc) (bf16, in place)
  softmax_kernel<<<dim3(BB * SS / 4), 128, 0, stream>>>(Sc, Pb);
  // out = P * V   (Bt = V^T rows, contiguous)
  gemm128_kernel<3><<<dim3(SS / 128, DD / 128, BB), 256, 0, stream>>>(
      Pb, ssBatch, VTb, sdBatch, nullptr, 1.f, d_out, sdBatch, SS);
}
